// PointNet2E2E_52544629899577
// MI455X (gfx1250) — matrix operand order repaired and checked by reading, not compiled, not run on hardware
//
#include <hip/hip_runtime.h>
#include <hip/hip_bf16.h>

typedef __attribute__((ext_vector_type(16))) _Float16 v16h;
typedef __attribute__((ext_vector_type(8)))  _Float16 v8h;
typedef __attribute__((ext_vector_type(8)))  float    v8f;
typedef __attribute__((ext_vector_type(4)))  unsigned int u32x4;
typedef __attribute__((ext_vector_type(8)))  int          i32x8;
typedef __attribute__((ext_vector_type(4)))  int          i32x4;

#define BN_SCALE_C 0.9999950000374997f   // 1/sqrt(1+1e-5)

#define USE_TDM (__has_builtin(__builtin_amdgcn_tensor_load_to_lds) && \
                 __has_builtin(__builtin_amdgcn_s_wait_tensorcnt))

// ---------------------------------------------------------------------------
// Tensor Data Mover: async 2D tile (64x64 f16) load global -> LDS.
// D# per ISA 08_async_tensor §8: group0 = {flags, lds_addr, global_addr, type},
// group1 = {data_size, tensor_dim0/1 (remaining extent -> OOB zero-fill),
//           tile_dim0/1 = 64, dim0 stride = row pitch}. 2D => groups 2/3 NULL.
// One DMA per instruction (EXEC ignored); completion via TENSORcnt.
// ---------------------------------------------------------------------------
#if USE_TDM
__device__ __forceinline__ void tdm_load_tile(_Float16* dstLds,
                                              const _Float16* src,
                                              unsigned rowsRem, unsigned colsRem,
                                              unsigned ldg) {
  unsigned long long ga = (unsigned long long)(size_t)src;
  u32x4 g0 = {};
  g0[0] = 1u;                                    // count=1, user descriptor
  g0[1] = (unsigned)(size_t)dstLds;              // lds_addr (LDS aperture low bits)
  g0[2] = (unsigned)ga;                          // global_addr[31:0]
  g0[3] = (unsigned)((ga >> 32) & 0x01FFFFFFull) | (2u << 30);  // addr hi | type=2
  i32x8 g1 = {};
  g1[0] = (int)(1u << 16);                       // data_size = 1 -> 2 bytes
  g1[1] = (int)((colsRem & 0xFFFFu) << 16);      // tensor_dim0[15:0]
  g1[2] = (int)(((colsRem >> 16) & 0xFFFFu) | ((rowsRem & 0xFFFFu) << 16));
  g1[3] = (int)(((rowsRem >> 16) & 0xFFFFu) | (64u << 16));     // tile_dim0 = 64
  g1[4] = (int)64;                               // tile_dim1 = 64, tile_dim2 = 0
  g1[5] = (int)ldg;                              // tensor_dim0_stride[31:0]
  g1[6] = 0;
  g1[7] = 0;
  i32x4 z4 = {};
#if __clang_major__ >= 23
  i32x8 z8 = {};
  __builtin_amdgcn_tensor_load_to_lds(g0, g1, z4, z4, z8, 0);
#else
  __builtin_amdgcn_tensor_load_to_lds(g0, g1, z4, z4, 0);
#endif
}
#endif

// ---------------------------------------------------------------------------
// WMMA fragment helpers (layouts per CDNA5 ISA 7.12.2, wave32)
// A 16x32 f16: lane l -> M = l&15 ; elements 0..7 = K[kb..kb+7],
//              elements 8..15 = K[kb+16..kb+23], kb = kk + 8*(l>>4)
// B 32x16 f16: lane l -> N = l&15 ; same K order as A
// C/D 16x16 f32: lane l, vgpr r -> M = r + 8*(l>>4), N = l&15
// ---------------------------------------------------------------------------
__device__ __forceinline__ v16h frag_a_lds(const _Float16* As, int lda, int m0,
                                           int kk, int lane) {
  int m  = m0 + (lane & 15);
  int kb = kk + ((lane >> 4) << 3);
  const _Float16* p = As + (size_t)m * lda + kb;
  v8h lo = *(const v8h*)p;
  v8h hi = *(const v8h*)(p + 16);
  return __builtin_shufflevector(lo, hi, 0,1,2,3,4,5,6,7,8,9,10,11,12,13,14,15);
}

__device__ __forceinline__ v16h frag_b_lds(const _Float16* Bs, int ldb, int kk,
                                           int n0, int lane) {
  int n  = n0 + (lane & 15);
  int kb = kk + ((lane >> 4) << 3);
  v16h r;
#pragma unroll
  for (int i = 0; i < 8; ++i) {
    r[i]     = Bs[(size_t)(kb + i) * ldb + n];
    r[8 + i] = Bs[(size_t)(kb + 16 + i) * ldb + n];
  }
  return r;
}

__device__ __forceinline__ v16h frag_b_gmem(const _Float16* W, int Nn, int Kr,
                                            int k0, int n0, int lane) {
  v16h z = {};
  int n  = n0 + (lane & 15);
  int kb = k0 + ((lane >> 4) << 3);
#pragma unroll
  for (int i = 0; i < 8; ++i) {
    int ka = kb + i, kc = kb + 16 + i;
    if (ka < Kr) z[i]     = W[(size_t)ka * Nn + n];   // zero rows for padded K
    if (kc < Kr) z[8 + i] = W[(size_t)kc * Nn + n];
  }
  return z;
}

// ---------------------------------------------------------------------------
// Weight folding: W' = W * g * BN_SCALE (per output col), b' = b*g*s + beta
// ---------------------------------------------------------------------------
__global__ void fold_layer_kernel(const float* __restrict__ W,
                                  const float* __restrict__ b,
                                  const float* __restrict__ g,
                                  const float* __restrict__ bt,
                                  int K, int N, _Float16* __restrict__ Wo,
                                  float* __restrict__ bo) {
  long total = (long)K * N;
  long i0 = (long)blockIdx.x * 256 + threadIdx.x;
  for (long i = i0; i < total; i += (long)gridDim.x * 256) {
    int n = (int)(i % N);
    float s = g ? g[n] * BN_SCALE_C : 1.f;
    Wo[i] = (_Float16)(W[i] * s);
  }
  if (i0 < N) {
    float s = g ? g[i0] * BN_SCALE_C : 1.f;
    bo[i0] = b[i0] * s + (bt ? bt[i0] : 0.f);
  }
}

// ---------------------------------------------------------------------------
// Farthest point sampling: one block per batch, dist[] lives in LDS
// ---------------------------------------------------------------------------
__global__ void __launch_bounds__(256)
fps_kernel(const float* __restrict__ xyz, int ldx, int N, int npoint,
           int* __restrict__ idx_out, float* __restrict__ new_xyz) {
  __shared__ float dist[3072];
  __shared__ float rv[256];
  __shared__ int   ri[256];
  __shared__ float cen[3];
  __shared__ int   sfar;
  const int tid = threadIdx.x, b = blockIdx.x;
  for (int i = tid; i < N; i += 256) dist[i] = 1e10f;
  if (tid == 0) sfar = 0;
  __syncthreads();
  for (int it = 0; it < npoint; ++it) {
    int far = sfar;
    if (tid == 0) {
      idx_out[b * npoint + it] = far;
      const float* q = xyz + ((size_t)b * N + far) * ldx;
      cen[0] = q[0]; cen[1] = q[1]; cen[2] = q[2];
      float* o = new_xyz + ((size_t)b * npoint + it) * 3;
      o[0] = q[0]; o[1] = q[1]; o[2] = q[2];
    }
    __syncthreads();
    float lm = -1.f; int la = 0;
    for (int i = tid; i < N; i += 256) {
      const float* q = xyz + ((size_t)b * N + i) * ldx;
      float dx = q[0] - cen[0], dy = q[1] - cen[1], dz = q[2] - cen[2];
      float d  = dx * dx + dy * dy + dz * dz;
      float nd = fminf(dist[i], d);
      dist[i] = nd;
      if (nd > lm) { lm = nd; la = i; }
    }
    rv[tid] = lm; ri[tid] = la;
    __syncthreads();
    for (int s = 128; s > 0; s >>= 1) {
      if (tid < s && rv[tid + s] > rv[tid]) { rv[tid] = rv[tid + s]; ri[tid] = ri[tid + s]; }
      __syncthreads();
    }
    if (tid == 0) sfar = ri[0];
    __syncthreads();
  }
}

// ---------------------------------------------------------------------------
// Fused SA stage: per-centroid ball query + gather + 3-layer WMMA MLP + maxpool
// ---------------------------------------------------------------------------
template <int NS>
__device__ __forceinline__ void mlp_layer(const _Float16* Ain, int Kp,
                                          const _Float16* Wg, int Kr, int Nn,
                                          const float* bias, _Float16* Aout) {
  const int tid = threadIdx.x, lane = tid & 31, wave = tid >> 5;
  const int MT = NS / 16, NT = Nn / 16;
  for (int t = wave; t < MT * NT; t += 4) {
    int mt = t % MT, nt = t / MT;
    v8f acc = {};
    for (int k0 = 0; k0 < Kp; k0 += 32) {
      v16h a = frag_a_lds(Ain, Kp, mt * 16, k0, lane);
      v16h w = frag_b_gmem(Wg, Nn, Kr, k0, nt * 16, lane);
      acc = __builtin_amdgcn_wmma_f32_16x16x32_f16(false, a, false, w,
                                                   (short)0, acc, false, false);
    }
    int n = nt * 16 + (lane & 15);
    float bv = bias[n];
#pragma unroll
    for (int r = 0; r < 8; ++r) {
      int m = mt * 16 + r + ((lane >> 4) << 3);
      float v = acc[r] + bv;
      v = v > 0.f ? v : 0.f;
      Aout[(size_t)m * Nn + n] = (_Float16)v;
    }
  }
}

template <int NS>
__device__ __forceinline__ void mlp_layer_max(const _Float16* Ain, int Kp,
                                              const _Float16* Wg, int Kr, int Nn,
                                              const float* bias, float* smax) {
  const int tid = threadIdx.x, lane = tid & 31, wave = tid >> 5;
  const int MT = NS / 16, NT = Nn / 16;
  for (int t = wave; t < MT * NT; t += 4) {
    int mt = t % MT, nt = t / MT;
    v8f acc = {};
    for (int k0 = 0; k0 < Kp; k0 += 32) {
      v16h a = frag_a_lds(Ain, Kp, mt * 16, k0, lane);
      v16h w = frag_b_gmem(Wg, Nn, Kr, k0, nt * 16, lane);
      acc = __builtin_amdgcn_wmma_f32_16x16x32_f16(false, a, false, w,
                                                   (short)0, acc, false, false);
    }
    int n = nt * 16 + (lane & 15);
    float bv = bias[n];
    float mx = 0.f;                       // ReLU floor
#pragma unroll
    for (int r = 0; r < 8; ++r) {
      float v = acc[r] + bv;
      v = v > 0.f ? v : 0.f;
      mx = fmaxf(mx, v);
    }
    mx = fmaxf(mx, __shfl_xor(mx, 16));   // combine the two 8-row halves
    if (lane < 16)
      atomicMax((int*)&smax[n], __float_as_int(mx));  // all values >= 0
  }
}

template <int NS, int KP, int CIN, int C1, int C2, int C3>
__global__ void __launch_bounds__(128)
sa_kernel(const float* __restrict__ sxyz, int ldsx,
          const float* __restrict__ sfeat, int ldf, int N,
          const float* __restrict__ nxyz, int npoint, float r2,
          const _Float16* __restrict__ W1, const float* __restrict__ b1,
          const _Float16* __restrict__ W2, const float* __restrict__ b2,
          const _Float16* __restrict__ W3, const float* __restrict__ b3,
          float* __restrict__ outFeat) {
  constexpr int KA = (KP > C2) ? KP : C2;
  __shared__ _Float16 bufA[NS * KA];
  __shared__ _Float16 bufB[NS * C1];
  __shared__ float    smax[C3];
  __shared__ int      nidx[NS];
  __shared__ float    cen[3];
  const int tid = threadIdx.x, lane = tid & 31, wave = tid >> 5;
  const int bp = blockIdx.x, b = bp / npoint;

  // speculative RT prefetch of the (L2-resident, heavily reused) weight tables
  __builtin_prefetch(W1 + (size_t)tid * 8, 0, 1);
  __builtin_prefetch(W2 + (size_t)tid * 8, 0, 1);
  __builtin_prefetch(W3 + (size_t)tid * 8, 0, 1);

  if (tid < 3) cen[tid] = nxyz[(size_t)bp * 3 + tid];
  for (int i = tid; i < C3; i += 128) smax[i] = 0.f;
  __syncthreads();

  if (wave == 0) {                         // ball query, ordered compaction
    int cnt = 0;
    for (int base = 0; base < N && cnt < NS; base += 32) {
      int j = base + lane;
      bool hit = false;
      if (j < N) {
        const float* q = sxyz + ((size_t)b * N + j) * ldsx;
        float dx = q[0] - cen[0], dy = q[1] - cen[1], dz = q[2] - cen[2];
        hit = (dx * dx + dy * dy + dz * dz) <= r2;
      }
#if __has_builtin(__builtin_amdgcn_ballot_w32)
      unsigned mask = __builtin_amdgcn_ballot_w32(hit);
#else
      unsigned mask = (unsigned)__ballot(hit);
#endif
      int pos = cnt + __popc(mask & ((1u << lane) - 1u));
      if (hit && pos < NS) nidx[pos] = j;
      cnt += __popc(mask);
    }
    if (cnt > NS) cnt = NS;
    if (lane == 0) {
      int f = (cnt > 0) ? nidx[0] : 0;
      for (int i = cnt; i < NS; ++i) nidx[i] = f;
    }
  }
  __syncthreads();

  for (int e = tid; e < NS * KP; e += 128) {  // gather, f32 -> f16, zero pad
    int j = e / KP, c = e % KP;
    int src = nidx[j];
    float v = 0.f;
    if (c < 3)        v = sxyz[((size_t)b * N + src) * ldsx + c] - cen[c];
    else if (c < CIN) v = sfeat[((size_t)b * N + src) * ldf + (c - 3)];
    bufA[(size_t)j * KP + c] = (_Float16)v;
  }
  __syncthreads();

  mlp_layer<NS>(bufA, KP, W1, CIN, C1, b1, bufB);
  __syncthreads();
  mlp_layer<NS>(bufB, C1, W2, C1, C2, b2, bufA);
  __syncthreads();
  mlp_layer_max<NS>(bufA, C2, W3, C2, C3, b3, smax);
  __syncthreads();

  for (int i = tid; i < C3; i += 128)
    outFeat[(size_t)bp * C3 + i] = smax[i];
}

// ---------------------------------------------------------------------------
// Generic WMMA GEMM: out = [relu](A[MxK](f16,lda) @ W[KxN](f16) + bias)
// 64x64 block tile, 8 waves, each wave one 16x32 accumulator pair.
// Tile staging goes through the Tensor Data Mover (TENSORcnt) when available.
// ---------------------------------------------------------------------------
__device__ __forceinline__ void store_tile(v8f c, int m0, int n0,
                                           const float* bias, float* outF,
                                           _Float16* outH, int M, int N,
                                           int relu, int lane) {
  int n = n0 + (lane & 15);
  if (n >= N) return;
  float bv = bias[n];
#pragma unroll
  for (int r = 0; r < 8; ++r) {
    int m = m0 + r + ((lane >> 4) << 3);
    if (m >= M) continue;
    float v = c[r] + bv;
    if (relu) v = v > 0.f ? v : 0.f;
    if (outF) outF[(size_t)m * N + n] = v;
    if (outH) outH[(size_t)m * N + n] = (_Float16)v;
  }
}

__global__ void __launch_bounds__(256)
gemm_kernel(const _Float16* __restrict__ A, int lda,
            const _Float16* __restrict__ W, const float* __restrict__ bias,
            float* __restrict__ outF, _Float16* __restrict__ outH,
            int M, int N, int K, int relu) {
  __shared__ _Float16 As[64 * 64];
  __shared__ _Float16 Bs[64 * 64];
  const int tid = threadIdx.x, lane = tid & 31, wave = tid >> 5;
  const int bm = blockIdx.y * 64, bn = blockIdx.x * 64;
  const int mt = wave & 3, nh = wave >> 2;
  v8f acc0 = {}, acc1 = {};

  for (int k0 = 0; k0 < K; k0 += 64) {
#if USE_TDM
    // Two async TDM tile DMAs (A and B); OOB rows/cols zero-fill in LDS.
    if (wave == 0) {
      tdm_load_tile(As, A + (size_t)bm * lda + k0,
                    (unsigned)(M - bm), (unsigned)(K - k0), (unsigned)lda);
      tdm_load_tile(Bs, W + (size_t)k0 * N + bn,
                    (unsigned)(K - k0), (unsigned)(N - bn), (unsigned)N);
    }
    if (k0 + 64 < K)   // speculative prefetch of next A chunk while DMA runs
      __builtin_prefetch(A + (size_t)(bm + (tid & 63)) * lda + k0 + 64, 0, 1);
    if (wave == 0) __builtin_amdgcn_s_wait_tensorcnt(0);
    __syncthreads();   // publishes the LDS tiles to all 8 waves
#else
    for (int i = tid; i < 512; i += 256) {            // stage A 64x64
      int r = i >> 3, c8 = (i & 7) << 3;
      int gm = bm + r, gk = k0 + c8;
      v8h v = {};
      if (gm < M) {
        if (gk + 8 <= K) v = *(const v8h*)(A + (size_t)gm * lda + gk);
        else for (int e = 0; e < 8; ++e)
          if (gk + e < K) v[e] = A[(size_t)gm * lda + gk + e];
      }
      *(v8h*)(As + r * 64 + c8) = v;
    }
    for (int i = tid; i < 512; i += 256) {            // stage B 64x64
      int r = i >> 3, c8 = (i & 7) << 3;
      int gk = k0 + r, gn = bn + c8;
      v8h v = {};
      if (gk < K) {
        if (gn + 8 <= N) v = *(const v8h*)(W + (size_t)gk * N + gn);
        else for (int e = 0; e < 8; ++e)
          if (gn + e < N) v[e] = W[(size_t)gk * N + gn + e];
      }
      *(v8h*)(Bs + r * 64 + c8) = v;
    }
    __syncthreads();
#endif
#pragma unroll
    for (int kk = 0; kk < 64; kk += 32) {
      v16h a  = frag_a_lds(As, 64, mt * 16, kk, lane);
      v16h b0 = frag_b_lds(Bs, 64, kk, nh * 32,      lane);
      v16h b1 = frag_b_lds(Bs, 64, kk, nh * 32 + 16, lane);
      acc0 = __builtin_amdgcn_wmma_f32_16x16x32_f16(false, a, false, b0,
                                                    (short)0, acc0, false, false);
      acc1 = __builtin_amdgcn_wmma_f32_16x16x32_f16(false, a, false, b1,
                                                    (short)0, acc1, false, false);
    }
    __syncthreads();
  }
  store_tile(acc0, bm + mt * 16, bn + nh * 32,      bias, outF, outH, M, N, relu, lane);
  store_tile(acc1, bm + mt * 16, bn + nh * 32 + 16, bias, outF, outH, M, N, relu, lane);
}

// ---------------------------------------------------------------------------
// 3-NN inverse-distance interpolation: one wave per target point
// ---------------------------------------------------------------------------
__global__ void __launch_bounds__(128)
interp3_kernel(const float* __restrict__ txyz, int ldt, int Nt,
               const float* __restrict__ sxyz, int S,
               const float* __restrict__ sfeat, int C,
               _Float16* __restrict__ dst, int ldd, int coff, int B) {
  const int tid = threadIdx.x, lane = tid & 31, wave = tid >> 5;
  const long g = (long)blockIdx.x * 4 + wave;
  if (g >= (long)B * Nt) return;
  const int b = (int)(g / Nt), i = (int)(g % Nt);
  const float* q = txyz + ((size_t)b * Nt + i) * ldt;
  float qx = q[0], qy = q[1], qz = q[2];
  int ch[3]; float cd[3];
  for (int t = 0; t < 3; ++t) {
    float bm = 3.0e38f; int ba = 0;
    for (int s = lane; s < S; s += 32) {
      bool skip = false;
      for (int u = 0; u < t; ++u) if (s == ch[u]) skip = true;
      if (skip) continue;
      const float* r = sxyz + ((size_t)b * S + s) * 3;
      float dx = r[0] - qx, dy = r[1] - qy, dz = r[2] - qz;
      float d  = dx * dx + dy * dy + dz * dz;
      if (d < bm) { bm = d; ba = s; }
    }
    for (int off = 16; off > 0; off >>= 1) {
      float ov = __shfl_xor(bm, off);
      int   oa = __shfl_xor(ba, off);
      if (ov < bm || (ov == bm && oa < ba)) { bm = ov; ba = oa; }
    }
    ch[t] = ba; cd[t] = bm;
  }
  float w0 = 1.f / (cd[0] + 1e-8f), w1 = 1.f / (cd[1] + 1e-8f),
        w2 = 1.f / (cd[2] + 1e-8f);
  float ws = w0 + w1 + w2;
  w0 /= ws; w1 /= ws; w2 /= ws;
  const float* f0 = sfeat + ((size_t)b * S + ch[0]) * C;
  const float* f1 = sfeat + ((size_t)b * S + ch[1]) * C;
  const float* f2 = sfeat + ((size_t)b * S + ch[2]) * C;
  _Float16* o = dst + ((size_t)b * Nt + i) * ldd + coff;
  for (int c = lane; c < C; c += 32)
    o[c] = (_Float16)(w0 * f0[c] + w1 * f1[c] + w2 * f2[c]);
}

// ---------------------------------------------------------------------------
// Pack / broadcast / row-max helpers
// ---------------------------------------------------------------------------
__global__ void pack_cols_kernel(const float* __restrict__ src, int Cs,
                                 _Float16* __restrict__ dst, int ldd, int coff,
                                 long M) {
  long total = M * Cs;
  for (long i = (long)blockIdx.x * 256 + threadIdx.x; i < total;
       i += (long)gridDim.x * 256) {
    long m = i / Cs; int c = (int)(i % Cs);
    dst[m * ldd + coff + c] = (_Float16)src[i];
  }
}

__global__ void pack_bcast_kernel(const float* __restrict__ src, int Cs,
                                  _Float16* __restrict__ dst, int ldd, int coff,
                                  int rows_per_b, long M) {
  long total = M * Cs;
  for (long i = (long)blockIdx.x * 256 + threadIdx.x; i < total;
       i += (long)gridDim.x * 256) {
    long m = i / Cs; int c = (int)(i % Cs);
    long b = m / rows_per_b;
    dst[m * ldd + coff + c] = (_Float16)src[b * Cs + c];
  }
}

__global__ void maxrows_kernel(const float* __restrict__ in, int R, int C,
                               float* __restrict__ out) {
  int b = blockIdx.y;
  int c = blockIdx.x * 256 + threadIdx.x;
  if (c >= C) return;
  float m = -3.0e38f;
  for (int r = 0; r < R; ++r)
    m = fmaxf(m, in[((size_t)b * R + r) * C + c]);
  out[(size_t)b * C + c] = m;
}

// ---------------------------------------------------------------------------
// Host orchestration
// ---------------------------------------------------------------------------
static void launch_gemm(const _Float16* A, int lda, const _Float16* W,
                        const float* bias, float* outF, _Float16* outH,
                        int M, int N, int K, int relu, hipStream_t s) {
  dim3 grid((N + 63) / 64, (M + 63) / 64);
  gemm_kernel<<<grid, 256, 0, s>>>(A, lda, W, bias, outF, outH, M, N, K, relu);
}

extern "C" void kernel_launch(void* const* d_in, const int* in_sizes, int n_in,
                              void* d_out, int out_size, void* d_ws, size_t ws_size,
                              hipStream_t stream) {
  (void)in_sizes; (void)n_in; (void)out_size; (void)ws_size;
  const float* xyzf = (const float*)d_in[0];

  struct LD { int widx, K, N, bn; };
  const LD L[18] = {
    {1, 16, 64, 1},  {5, 64, 64, 1},   {9, 64, 128, 1},     // sa1
    {13, 131, 128, 1},{17, 128, 128, 1},{21, 128, 256, 1},  // sa2
    {25, 259, 256, 1},{29, 256, 512, 1},{33, 512, 1024, 1}, // sa3
    {37, 1280, 512, 1},{41, 512, 256, 1},                   // fp3
    {45, 384, 256, 1},{49, 256, 128, 1},                    // fp2
    {53, 141, 128, 1},{57, 128, 64, 1},{61, 64, 32, 1},     // fp1
    {65, 32, 16, 1},                                        // out1
    {69, 16, 3, 0},                                         // out2
  };

  char* base = (char*)d_ws; size_t off = 0;
  auto alloc = [&](size_t bytes) -> void* {
    void* p = base + off;
    off += (bytes + 255) & ~(size_t)255;
    return p;
  };

  _Float16* Wf[18]; float* Bf[18];
  for (int l = 0; l < 18; ++l) {
    Wf[l] = (_Float16*)alloc((size_t)L[l].K * L[l].N * 2);
    Bf[l] = (float*)alloc((size_t)L[l].N * 4);
  }
  const int B = 16, N0 = 3000;
  int*   idx1  = (int*)alloc((size_t)B * 1024 * 4);
  float* nxyz1 = (float*)alloc((size_t)B * 1024 * 3 * 4);
  float* l1    = (float*)alloc((size_t)B * 1024 * 128 * 4);
  int*   idx2  = (int*)alloc((size_t)B * 128 * 4);
  float* nxyz2 = (float*)alloc((size_t)B * 128 * 3 * 4);
  float* l2    = (float*)alloc((size_t)B * 128 * 256 * 4);
  _Float16* A3    = (_Float16*)alloc((size_t)2048 * 264 * 2);
  _Float16* t3a   = (_Float16*)alloc((size_t)2048 * 256 * 2);
  _Float16* t3b   = (_Float16*)alloc((size_t)2048 * 512 * 2);
  float*    sa3o  = (float*)alloc((size_t)2048 * 1024 * 4);
  float*    l3    = (float*)alloc((size_t)B * 1024 * 4);
  _Float16* Afp3  = (_Float16*)alloc((size_t)2048 * 1280 * 2);
  _Float16* tfp3  = (_Float16*)alloc((size_t)2048 * 512 * 2);
  float*    l2p   = (float*)alloc((size_t)2048 * 256 * 4);
  _Float16* Afp2  = (_Float16*)alloc((size_t)16384 * 384 * 2);
  _Float16* tfp2  = (_Float16*)alloc((size_t)16384 * 256 * 2);
  float*    l1p   = (float*)alloc((size_t)16384 * 128 * 4);
  _Float16* Afp1  = (_Float16*)alloc((size_t)48000 * 144 * 2);
  _Float16* t1a   = (_Float16*)alloc((size_t)48000 * 128 * 2);
  _Float16* t1b   = (_Float16*)alloc((size_t)48000 * 64 * 2);
  _Float16* l0h   = (_Float16*)alloc((size_t)48000 * 32 * 2);
  _Float16* h16   = (_Float16*)alloc((size_t)48000 * 16 * 2);

  // 1) fold BN into weights, convert to f16
  for (int l = 0; l < 18; ++l) {
    const float* W  = (const float*)d_in[L[l].widx];
    const float* bb = (const float*)d_in[L[l].widx + 1];
    const float* g  = L[l].bn ? (const float*)d_in[L[l].widx + 2] : nullptr;
    const float* bt = L[l].bn ? (const float*)d_in[L[l].widx + 3] : nullptr;
    long total = (long)L[l].K * L[l].N;
    int grid = (int)((total + 255) / 256); if (grid > 1024) grid = 1024;
    fold_layer_kernel<<<grid, 256, 0, stream>>>(W, bb, g, bt, L[l].K, L[l].N,
                                                Wf[l], Bf[l]);
  }

  // 2) SA1: FPS(3000->1024) + fused group MLP [16,64,64,128], r=0.2, ns=32
  fps_kernel<<<B, 256, 0, stream>>>(xyzf, 13, N0, 1024, idx1, nxyz1);
  sa_kernel<32, 32, 16, 64, 64, 128><<<B * 1024, 128, 0, stream>>>(
      xyzf, 13, xyzf, 13, N0, nxyz1, 1024, 0.04f,
      Wf[0], Bf[0], Wf[1], Bf[1], Wf[2], Bf[2], l1);

  // 3) SA2: FPS(1024->128) + fused group MLP [131,128,128,256], r=0.4, ns=64
  fps_kernel<<<B, 256, 0, stream>>>(nxyz1, 3, 1024, 128, idx2, nxyz2);
  sa_kernel<64, 160, 131, 128, 128, 256><<<B * 128, 128, 0, stream>>>(
      nxyz1, 3, l1, 128, 1024, nxyz2, 128, 0.16f,
      Wf[3], Bf[3], Wf[4], Bf[4], Wf[5], Bf[5], l2);

  // 4) SA3 global: concat(xyz2, l2) -> [259,256,512,1024] -> max over 128 pts
  const long M3 = 2048;
  pack_cols_kernel<<<1024, 256, 0, stream>>>(nxyz2, 3, A3, 264, 0, M3);
  pack_cols_kernel<<<1024, 256, 0, stream>>>(l2, 256, A3, 264, 3, M3);
  launch_gemm(A3, 264, Wf[6], Bf[6], nullptr, t3a, 2048, 256, 259, 1, stream);
  launch_gemm(t3a, 256, Wf[7], Bf[7], nullptr, t3b, 2048, 512, 256, 1, stream);
  launch_gemm(t3b, 512, Wf[8], Bf[8], sa3o, nullptr, 2048, 1024, 512, 1, stream);
  maxrows_kernel<<<dim3(4, B), 256, 0, stream>>>(sa3o, 128, 1024, l3);

  // 5) FP3: concat(l2, bcast l3) -> [1280,512,256]
  pack_cols_kernel<<<1024, 256, 0, stream>>>(l2, 256, Afp3, 1280, 0, M3);
  pack_bcast_kernel<<<1024, 256, 0, stream>>>(l3, 1024, Afp3, 1280, 256, 128, M3);
  launch_gemm(Afp3, 1280, Wf[9], Bf[9], nullptr, tfp3, 2048, 512, 1280, 1, stream);
  launch_gemm(tfp3, 512, Wf[10], Bf[10], l2p, nullptr, 2048, 256, 512, 1, stream);

  // 6) FP2: concat(l1, interp3(l2p)) -> [384,256,128]
  const long M2 = (long)B * 1024;
  pack_cols_kernel<<<1024, 256, 0, stream>>>(l1, 128, Afp2, 384, 0, M2);
  interp3_kernel<<<(B * 1024 + 3) / 4, 128, 0, stream>>>(
      nxyz1, 3, 1024, nxyz2, 128, l2p, 256, Afp2, 384, 128, B);
  launch_gemm(Afp2, 384, Wf[11], Bf[11], nullptr, tfp2, 16384, 256, 384, 1, stream);
  launch_gemm(tfp2, 256, Wf[12], Bf[12], l1p, nullptr, 16384, 128, 256, 1, stream);

  // 7) FP1: concat(x, interp3(l1p)) -> [141,128,64,32]
  const long M1 = (long)B * N0;
  pack_cols_kernel<<<1024, 256, 0, stream>>>(xyzf, 13, Afp1, 144, 0, M1);
  interp3_kernel<<<(B * N0 + 3) / 4, 128, 0, stream>>>(
      xyzf, 13, N0, nxyz1, 1024, l1p, 128, Afp1, 144, 13, B);
  launch_gemm(Afp1, 144, Wf[13], Bf[13], nullptr, t1a, 48000, 128, 141, 1, stream);
  launch_gemm(t1a, 128, Wf[14], Bf[14], nullptr, t1b, 48000, 64, 128, 1, stream);
  launch_gemm(t1b, 64, Wf[15], Bf[15], nullptr, l0h, 48000, 32, 64, 1, stream);

  // 8) heads: 32->16 (BN+ReLU), 16->3 (linear) -> d_out
  launch_gemm(l0h, 32, Wf[16], Bf[16], nullptr, h16, 48000, 16, 32, 1, stream);
  launch_gemm(h16, 16, Wf[17], Bf[17], (float*)d_out, nullptr, 48000, 3, 16, 0, stream);
}
